// GCN_Dropout_28243704939123
// MI455X (gfx1250) — matrix operand order repaired and checked by reading, not compiled, not run on hardware
//
#include <hip/hip_runtime.h>
#include <stdint.h>

// ---------------------------------------------------------------------------
// GCN 3-layer forward for MI455X (gfx1250, wave32).
//  - GEMM on WMMA bf16 (16x16x32), bf16x3 split for ~fp32 accuracy
//  - weights pre-split/packed into B-fragment order (no in-loop packing VALU)
//  - A panel staged via GLOBAL_LOAD_ASYNC_TO_LDS_B128 double-buffer (ASYNCcnt)
//  - edge scatter via global_atomic_add_f32 (working set is L2-resident)
//  - wave-per-row ReLU+LayerNorm with shfl_xor reduction (wave32)
// ---------------------------------------------------------------------------

#define N_NODES 50000
#define N_EDGES 800000
#define LN_EPS 1e-5f
#define PANELS 5            // 16-row panels per block -> 80 rows, grid = 625

typedef __bf16 bf16_t;
typedef bf16_t   v16bf __attribute__((ext_vector_type(16)));
typedef float    v8f   __attribute__((ext_vector_type(8)));
typedef uint32_t u32x4 __attribute__((ext_vector_type(4)));
typedef int      i32x4_vs __attribute__((vector_size(16)));   // builtin pointee

union Frag16 { v16bf v; u32x4 q[2]; uint32_t w[8]; };
union Acc8   { v8f v;   float f[8]; };

#if __has_builtin(__builtin_amdgcn_global_load_async_to_lds_b128)
#define HAVE_ASYNC_LDS 1
#else
#define HAVE_ASYNC_LDS 0
#endif

#if HAVE_ASYNC_LDS
__device__ __forceinline__ void async_copy_b128(const float* g, float* l) {
    // generic->AS1 keeps the address; generic LDS addr low 32 bits == LDS offset
    __builtin_amdgcn_global_load_async_to_lds_b128(
        (__attribute__((address_space(1))) i32x4_vs*)(uintptr_t)g,
        (__attribute__((address_space(3))) i32x4_vs*)(uint32_t)(uintptr_t)l,
        0, 0);
}
__device__ __forceinline__ void wait_async0() {
#if __has_builtin(__builtin_amdgcn_s_wait_asynccnt)
    __builtin_amdgcn_s_wait_asynccnt(0);
#else
    asm volatile("s_wait_asynccnt 0" ::: "memory");
#endif
}
#endif

__device__ __forceinline__ uint32_t f32_to_bf16_rne(float x) {
    uint32_t u = __float_as_uint(x);
    return (u + 0x7FFFu + ((u >> 16) & 1u)) >> 16;
}
__device__ __forceinline__ float bf16_bits_to_f32(uint32_t h) {
    return __uint_as_float(h << 16);
}
__device__ __forceinline__ void split_bf16(float x, uint32_t& hi, uint32_t& lo) {
    hi = f32_to_bf16_rne(x);
    lo = f32_to_bf16_rne(x - bf16_bits_to_f32(hi));
}

// ---------------------------------------------------------------------------
// Degree / normalization (deg = 1 self loop + in-degree; shared by layers)
// ---------------------------------------------------------------------------
__global__ void gcn_deg_init(float* deg) {
    int i = blockIdx.x * blockDim.x + threadIdx.x;
    if (i < N_NODES) deg[i] = 1.0f;
}
__global__ void gcn_deg_count(const int* __restrict__ dst, float* deg) {
    int e = blockIdx.x * blockDim.x + threadIdx.x;
    if (e < N_EDGES) atomicAdd(&deg[dst[e]], 1.0f);
}
__global__ void gcn_dinv(float* deg) {
    int i = blockIdx.x * blockDim.x + threadIdx.x;
    if (i < N_NODES) deg[i] = rsqrtf(deg[i]);   // deg >= 1 always
}

// ---------------------------------------------------------------------------
// Pre-split W[K,M] into hi/lo bf16 packed in B-fragment word order:
//   word index = (((k0/32)*2 + half)*M + n)*8 + i ,  k = k0 + 16*half + 2*i
// so each lane's 8-word fragment is 32 contiguous bytes (2 x b128 loads).
// ---------------------------------------------------------------------------
template <int K, int M>
__global__ void gcn_pack_w(const float* __restrict__ W,
                           uint32_t* __restrict__ Hi, uint32_t* __restrict__ Lo) {
    int widx = blockIdx.x * blockDim.x + threadIdx.x;
    if (widx >= (K / 2) * M) return;
    int i    = widx & 7;
    int rest = widx >> 3;
    int n    = rest % M;  rest /= M;
    int half = rest & 1;
    int ks   = rest >> 1;
    int k    = ks * 32 + 16 * half + 2 * i;
    uint32_t h0, l0, h1, l1;
    split_bf16(W[(size_t)k * M + n],       h0, l0);
    split_bf16(W[(size_t)(k + 1) * M + n], h1, l1);
    Hi[widx] = h0 | (h1 << 16);
    Lo[widx] = l0 | (l1 << 16);
}

// ---------------------------------------------------------------------------
// GEMM: C[N,M] = A[N,K] @ W[K,M] with pre-packed split weights.
// Block = 256 threads (8 waves), PANELS x 16-row panels per block.
// A panel async-staged to LDS (f32, double buffer), split to bf16 hi/lo once,
// then each wave computes 16x16 col-tiles with 3 WMMAs per 32-k step.
// ---------------------------------------------------------------------------
template <int K, int M>
__global__ __launch_bounds__(256)
void gcn_gemm_wmma(const float* __restrict__ A,
                   const uint32_t* __restrict__ WpHi,
                   const uint32_t* __restrict__ WpLo,
                   float* __restrict__ C) {
#if HAVE_ASYNC_LDS
    __shared__ __align__(16) float sF[2][16][K];     // raw f32 panels (async dst)
#endif
    __shared__ __align__(16) uint16_t sHi[16][K + 8];
    __shared__ __align__(16) uint16_t sLo[16][K + 8];

    const int tid  = threadIdx.x;
    const int wave = tid >> 5;
    const int lane = tid & 31;
    const int half = lane >> 4;
    const int lm   = lane & 15;
    const int blkRow0 = blockIdx.x * (16 * PANELS);

#if HAVE_ASYNC_LDS
    // issue async copy of panel 0
    {
        #pragma unroll
        for (int t = 0; t < (K / 64); t++) {         // (16*K/4) b128 ops / 256 thr
            int e = tid + t * 256;                   // float4 index
            int r = e / (K / 4), c4 = e % (K / 4);
            async_copy_b128(&A[(size_t)(blkRow0 + r) * K + c4 * 4],
                            &sF[0][r][c4 * 4]);
        }
    }
#endif

    for (int p = 0; p < PANELS; ++p) {
        const int row0 = blkRow0 + p * 16;

#if HAVE_ASYNC_LDS
        wait_async0();
        __syncthreads();
        const int buf = p & 1;
        // split staged f32 panel -> bf16 hi/lo
        for (int idx = tid; idx < 16 * K; idx += 256) {
            int r = idx / K, c = idx % K;
            uint32_t hi, lo; split_bf16(sF[buf][r][c], hi, lo);
            sHi[r][c] = (uint16_t)hi;
            sLo[r][c] = (uint16_t)lo;
        }
        __syncthreads();
        if (p + 1 < PANELS) {                        // overlap next copy w/ math
            const int nb = (p + 1) & 1;
            const int nrow0 = blkRow0 + (p + 1) * 16;
            #pragma unroll
            for (int t = 0; t < (K / 64); t++) {
                int e = tid + t * 256;
                int r = e / (K / 4), c4 = e % (K / 4);
                async_copy_b128(&A[(size_t)(nrow0 + r) * K + c4 * 4],
                                &sF[nb][r][c4 * 4]);
            }
        }
#else
        for (int idx = tid; idx < 16 * K; idx += 256) {
            int r = idx / K, c = idx % K;
            uint32_t hi, lo; split_bf16(A[(size_t)(row0 + r) * K + c], hi, lo);
            sHi[r][c] = (uint16_t)hi;
            sLo[r][c] = (uint16_t)lo;
        }
        __syncthreads();
#endif

        for (int ct = wave; ct < (M >> 4); ct += 8) {
            Acc8 acc;
            #pragma unroll
            for (int r = 0; r < 8; r++) acc.f[r] = 0.0f;
            const int n = (ct << 4) + lm;

            #pragma unroll
            for (int k0 = 0; k0 < K; k0 += 32) {
                // A fragments: contiguous k = 16*half + s (matches packed B) -> one 32B run
                Frag16 aHi, aLo;
                {
                    const uint16_t* rHi = &sHi[lm][0];
                    const uint16_t* rLo = &sLo[lm][0];
                    const int kb = k0 + 16 * half;
                    aHi.q[0] = *(const u32x4*)&rHi[kb];
                    aHi.q[1] = *(const u32x4*)&rHi[kb + 8];
                    aLo.q[0] = *(const u32x4*)&rLo[kb];
                    aLo.q[1] = *(const u32x4*)&rLo[kb + 8];
                }
                // B fragments: 32 contiguous bytes per lane (pre-packed)
                Frag16 bHi, bLo;
                {
                    const size_t base =
                        ((size_t)(((k0 >> 5) << 1) + half) * M + n) << 3;
                    bHi.q[0] = *(const u32x4*)&WpHi[base];
                    bHi.q[1] = *(const u32x4*)&WpHi[base + 4];
                    bLo.q[0] = *(const u32x4*)&WpLo[base];
                    bLo.q[1] = *(const u32x4*)&WpLo[base + 4];
                }
                // bf16x3: A*B ~= Ahi*Bhi + Ahi*Blo + Alo*Bhi (f32 accumulate)
                acc.v = __builtin_amdgcn_wmma_f32_16x16x32_bf16(
                            false, aHi.v, false, bHi.v, (short)0, acc.v, false, false);
                acc.v = __builtin_amdgcn_wmma_f32_16x16x32_bf16(
                            false, aHi.v, false, bLo.v, (short)0, acc.v, false, false);
                acc.v = __builtin_amdgcn_wmma_f32_16x16x32_bf16(
                            false, aLo.v, false, bHi.v, (short)0, acc.v, false, false);
            }
            // C/D layout: VGPR r -> row r + 8*half, col = lane%16
            #pragma unroll
            for (int r = 0; r < 8; r++) {
                C[(size_t)(row0 + r + 8 * half) * M + n] = acc.f[r];
            }
        }
        __syncthreads();   // protect sHi/sLo before next panel's split
    }
}

// ---------------------------------------------------------------------------
// Aggregation: out[i] = bias + dinv[i]^2 * h[i] (self loop), then per edge
// out[dst] += dinv[src]*dinv[dst]*h[src] via f32 atomics (resolve in L2).
// ---------------------------------------------------------------------------
template <int M>
__global__ void gcn_agg_init(const float* __restrict__ h,
                             const float* __restrict__ bias,
                             const float* __restrict__ dinv,
                             float* __restrict__ out) {
    int i = blockIdx.x * blockDim.x + threadIdx.x;
    if (i >= N_NODES * M) return;
    int node = i / M, f = i % M;
    float d = dinv[node];
    out[i] = bias[f] + d * d * h[i];
}

template <int M>
__global__ __launch_bounds__(256)
void gcn_agg_edges(const float* __restrict__ h,
                   const int* __restrict__ src, const int* __restrict__ dst,
                   const float* __restrict__ dinv, float* __restrict__ out) {
    int lane = threadIdx.x & 31;
    int e = (blockIdx.x * blockDim.x + threadIdx.x) >> 5;  // one wave32 / edge
    if (e >= N_EDGES) return;
    int s = src[e], d = dst[e];
    float w = dinv[s] * dinv[d];
    const float* hs = h + (size_t)s * M;
    float* od = out + (size_t)d * M;
    #pragma unroll
    for (int j = 0; j < (M >> 5); j++) {
        int f = lane + (j << 5);
        atomicAdd(&od[f], w * hs[f]);
    }
}

// ---------------------------------------------------------------------------
// Fused ReLU + LayerNorm, one wave32 per row, shfl_xor reduction.
// ---------------------------------------------------------------------------
template <int M>
__global__ __launch_bounds__(256)
void gcn_relu_ln(const float* __restrict__ in, const float* __restrict__ g,
                 const float* __restrict__ be, float* __restrict__ out) {
    int lane = threadIdx.x & 31;
    int row = (blockIdx.x * blockDim.x + threadIdx.x) >> 5;
    if (row >= N_NODES) return;
    const float* xi = in + (size_t)row * M;
    float r[M >> 5];
    float s1 = 0.0f, s2 = 0.0f;
    #pragma unroll
    for (int j = 0; j < (M >> 5); j++) {
        float v = fmaxf(xi[lane + (j << 5)], 0.0f);
        r[j] = v; s1 += v; s2 += v * v;
    }
    #pragma unroll
    for (int o = 16; o > 0; o >>= 1) {
        s1 += __shfl_xor(s1, o, 32);
        s2 += __shfl_xor(s2, o, 32);
    }
    float mu  = s1 * (1.0f / (float)M);
    float var = s2 * (1.0f / (float)M) - mu * mu;
    float inv = rsqrtf(var + LN_EPS);
    float* yo = out + (size_t)row * M;
    #pragma unroll
    for (int j = 0; j < (M >> 5); j++) {
        int f = lane + (j << 5);
        yo[f] = (r[j] - mu) * inv * g[f] + be[f];
    }
}

// ---------------------------------------------------------------------------
extern "C" void kernel_launch(void* const* d_in, const int* in_sizes, int n_in,
                              void* d_out, int out_size, void* d_ws, size_t ws_size,
                              hipStream_t stream) {
    const float* x   = (const float*)d_in[0];
    const int*   ei  = (const int*)  d_in[1];
    const float* W1  = (const float*)d_in[2];
    const float* b1  = (const float*)d_in[3];
    const float* g1  = (const float*)d_in[4];
    const float* be1 = (const float*)d_in[5];
    const float* W2  = (const float*)d_in[6];
    const float* b2  = (const float*)d_in[7];
    const float* g2  = (const float*)d_in[8];
    const float* be2 = (const float*)d_in[9];
    const float* W3  = (const float*)d_in[10];
    const float* b3  = (const float*)d_in[11];
    const float* g3  = (const float*)d_in[12];
    const float* be3 = (const float*)d_in[13];
    float* out = (float*)d_out;

    const int* src = ei;
    const int* dst = ei + N_EDGES;

    // workspace: dinv[N] | WpHi[32K u32] | WpLo[32K u32] | bufA|bufB|bufC [N*256]
    char* ws = (char*)d_ws;
    float* dinv = (float*)ws;
    size_t off = ((size_t)N_NODES * 4 + 255) & ~(size_t)255;
    uint32_t* wpHi = (uint32_t*)(ws + off); off += (size_t)32768 * 4;
    uint32_t* wpLo = (uint32_t*)(ws + off); off += (size_t)32768 * 4;
    float* bufA = (float*)(ws + off);
    float* bufB = bufA + (size_t)N_NODES * 256;
    float* bufC = bufB + (size_t)N_NODES * 256;

    const int T = 256;
    dim3 blk(T);
    int gN  = (N_NODES + T - 1) / T;
    int gE  = (N_EDGES + T - 1) / T;
    int gEW = (int)(((size_t)N_EDGES * 32 + T - 1) / T);   // wave per edge
    int gRW = (int)(((size_t)N_NODES * 32 + T - 1) / T);   // wave per row
    int gNM    = (int)(((size_t)N_NODES * 256 + T - 1) / T);
    int gNM128 = (int)(((size_t)N_NODES * 128 + T - 1) / T);
    int gGemm  = N_NODES / (16 * PANELS);                  // 625, exact
    int gPack256x256 = (256 * 256 / 2 + T - 1) / T;
    int gPack128     = (128 * 256 / 2 + T - 1) / T;

    // normalization (once; shared across layers)
    gcn_deg_init<<<gN, blk, 0, stream>>>(dinv);
    gcn_deg_count<<<gE, blk, 0, stream>>>(dst, dinv);
    gcn_dinv<<<gN, blk, 0, stream>>>(dinv);

    // ---- layer 1: x[N,128] @ W1[128,256] ----
    gcn_pack_w<128, 256><<<gPack128, blk, 0, stream>>>(W1, wpHi, wpLo);
    gcn_gemm_wmma<128, 256><<<gGemm, blk, 0, stream>>>(x, wpHi, wpLo, bufB);
    gcn_agg_init<256><<<gNM, blk, 0, stream>>>(bufB, b1, dinv, bufC);
    gcn_agg_edges<256><<<gEW, blk, 0, stream>>>(bufB, src, dst, dinv, bufC);
    gcn_relu_ln<256><<<gRW, blk, 0, stream>>>(bufC, g1, be1, bufA);

    // ---- layer 2: A[N,256] @ W2[256,256] ----
    gcn_pack_w<256, 256><<<gPack256x256, blk, 0, stream>>>(W2, wpHi, wpLo);
    gcn_gemm_wmma<256, 256><<<gGemm, blk, 0, stream>>>(bufA, wpHi, wpLo, bufB);
    gcn_agg_init<256><<<gNM, blk, 0, stream>>>(bufB, b2, dinv, bufC);
    gcn_agg_edges<256><<<gEW, blk, 0, stream>>>(bufB, src, dst, dinv, bufC);
    gcn_relu_ln<256><<<gRW, blk, 0, stream>>>(bufC, g2, be2, bufA);

    // ---- layer 3: A[N,256] @ W3[256,128] ----
    gcn_pack_w<256, 128><<<gPack128, blk, 0, stream>>>(W3, wpHi, wpLo);
    gcn_gemm_wmma<256, 128><<<gGemm, blk, 0, stream>>>(bufA, wpHi, wpLo, bufB);
    gcn_agg_init<128><<<gNM128, blk, 0, stream>>>(bufB, b3, dinv, bufC);
    gcn_agg_edges<128><<<gEW, blk, 0, stream>>>(bufB, src, dst, dinv, bufC);
    gcn_relu_ln<128><<<gRW, blk, 0, stream>>>(bufC, g3, be3, out);
}
